// Model_14482629722746
// MI455X (gfx1250) — compile-verified, matrix operand order checked by reading, not run on hardware
//
#include <hip/hip_runtime.h>
#include <hip/hip_bf16.h>

// ---------------- model dims ----------------
#define BQ       8
#define LQ       512
#define ENC_IN   7
#define D_MODEL  768
#define D_INNER  1536
#define DT_RANK  48
#define DT_PAD   64          // dt_proj K padded to WMMA tile
#define N_STATE  16
#define K_CONV   4
#define E_LAYERS 2
#define C_OUT    7
#define PRED_LEN 96
#define ROWS     (BQ * LQ)   // 4096
#define XD       (DT_RANK + 2 * N_STATE)   // 80
#define XD_PAD   128         // x_proj N padded for tile loads

typedef __attribute__((ext_vector_type(16))) __bf16 v16bf;
typedef __attribute__((ext_vector_type(8)))  float  v8f;
typedef unsigned short u16;
typedef unsigned int   u32;

static __device__ __forceinline__ u16 f2bf(float f) {
    u32 u = __float_as_uint(f);
    u += 0x7FFFu + ((u >> 16) & 1u);           // round-to-nearest-even
    return (u16)(u >> 16);
}
static __device__ __forceinline__ float sigm(float v) {
    return 1.0f / (1.0f + __expf(-v));
}
// async copy 16 bytes global -> LDS (ASYNCcnt-tracked, bypasses VGPRs)
static __device__ __forceinline__ void async_cp16(u32 lds_byte_off, const void* gptr) {
    asm volatile("global_load_async_to_lds_b128 %0, %1, off"
                 :: "v"(lds_byte_off), "v"(gptr) : "memory");
}
static __device__ __forceinline__ u32 lds_off(const void* p) {
    return (u32)(unsigned long long)p;         // LDS generic addr: low 32 bits = offset
}

// ============================================================
// 0a) f32 -> bf16 elementwise conversion
// ============================================================
__global__ void cvt_bf16_kernel(const float* __restrict__ in,
                                u16* __restrict__ out, int n) {
    int i = blockIdx.x * 256 + threadIdx.x;
    if (i < n) out[i] = f2bf(in[i]);
}

// ============================================================
// 0b) f32 -> bf16 with zero padding: [rin][cin] -> [rout][cout]
// ============================================================
__global__ void cvt_pad_bf16_kernel(const float* __restrict__ in,
                                    u16* __restrict__ out,
                                    int rin, int cin, int rout, int cout) {
    int i = blockIdx.x * 256 + threadIdx.x;
    if (i >= rout * cout) return;
    int r = i / cout, c = i % cout;
    out[i] = (r < rin && c < cin) ? f2bf(in[(size_t)r * cin + c]) : (u16)0;
}

// ============================================================
// 1) per-(b,c) mean / std over L
// ============================================================
__global__ void stats_kernel(const float* __restrict__ x,
                             float* __restrict__ mean_s, float* __restrict__ std_s) {
    __shared__ float r0[256], r1[256];
    int bc = blockIdx.x;
    int b = bc / ENC_IN, c = bc % ENC_IN;
    int tid = threadIdx.x;
    float s = 0.f, sq = 0.f;
    for (int l = tid; l < LQ; l += 256) {
        float v = x[((size_t)(b * LQ + l)) * ENC_IN + c];
        s += v; sq += v * v;
    }
    r0[tid] = s; r1[tid] = sq; __syncthreads();
    for (int st = 128; st > 0; st >>= 1) {
        if (tid < st) { r0[tid] += r0[tid + st]; r1[tid] += r1[tid + st]; }
        __syncthreads();
    }
    if (tid == 0) {
        float m = r0[0] / (float)LQ;
        float var = r1[0] / (float)LQ - m * m;
        mean_s[bc] = m;
        std_s[bc]  = sqrtf(var + 1e-5f);
    }
}

// ============================================================
// 2) token conv (circular pad) + positional + time embedding
// ============================================================
__global__ void embed_kernel(const float* __restrict__ x,
                             const float* __restrict__ tw,
                             const float* __restrict__ time_w,
                             const float* __restrict__ mean_s,
                             const float* __restrict__ std_s,
                             float* __restrict__ h) {
    int idx = blockIdx.x * 256 + threadIdx.x;
    if (idx >= ROWS * D_MODEL) return;
    int row = idx / D_MODEL, d = idx % D_MODEL;
    int b = row / LQ, l = row % LQ;
    float acc = 0.f;
#pragma unroll
    for (int k = 0; k < 3; ++k) {
        int ls = (l - 1 + k + LQ) % LQ;
        const float* xrow = x + ((size_t)(b * LQ + ls)) * ENC_IN;
#pragma unroll
        for (int c = 0; c < ENC_IN; ++c) {
            float xc = (xrow[c] - mean_s[b * ENC_IN + c]) / std_s[b * ENC_IN + c];
            acc = fmaf(tw[(d * ENC_IN + c) * 3 + k], xc, acc);
        }
    }
    int half = d >> 1;
    float freq = __expf(-(float)(2 * half) * (logf(10000.f) / (float)D_MODEL));
    float ang = (float)l * freq;
    float pe = (d & 1) ? __cosf(ang) : __sinf(ang);
    float tm = ((float)l / (float)LQ) * time_w[d];
    h[idx] = acc + pe + tm;
}

// ============================================================
// 3) rmsnorm over last dim -> bf16 output (feeds WMMA GEMM)
// ============================================================
__global__ void rmsnorm_kernel(const float* __restrict__ xin,
                               const float* __restrict__ w,
                               u16* __restrict__ xout) {
    __shared__ float red[256];
    __shared__ float scale_s;
    int row = blockIdx.x, tid = threadIdx.x;
    const float* xi = xin + (size_t)row * D_MODEL;
    float ss = 0.f;
    for (int d = tid; d < D_MODEL; d += 256) { float v = xi[d]; ss += v * v; }
    red[tid] = ss; __syncthreads();
    for (int st = 128; st > 0; st >>= 1) {
        if (tid < st) red[tid] += red[tid + st];
        __syncthreads();
    }
    if (tid == 0) scale_s = rsqrtf(red[0] / (float)D_MODEL + 1e-5f);
    __syncthreads();
    float sc = scale_s;
    u16* xo = xout + (size_t)row * D_MODEL;
    for (int d = tid; d < D_MODEL; d += 256) xo[d] = f2bf(xi[d] * sc * w[d]);
}

// ============================================================
// 4) WMMA GEMM (bf16 in, f32 out):  C[M,N] (+)= A[M,K] * W[N,K]^T
//    block 256 (8 waves), tile 128x64, K step 64 (8 WMMA / stage)
//    double-buffered GLOBAL_LOAD_ASYNC_TO_LDS_B128 pipeline.
//    Caller guarantees: M % 128 == 0, K % 64 == 0, and B rows
//    [bN, bN+64) are loadable (padded buffers). Store guards N.
// ============================================================
__global__ __launch_bounds__(256)
void gemm_bf16_wmma(const u16* __restrict__ A, int lda,
                    const u16* __restrict__ Bw, int ldb,
                    float* __restrict__ C, int ldc,
                    int M, int N, int K, int addC) {
    __shared__ __attribute__((aligned(128))) u16 sA[2][128 * 64];
    __shared__ __attribute__((aligned(128))) u16 sB[2][64 * 64];

    int tid  = threadIdx.x;
    int lane = tid & 31;
    int wave = tid >> 5;
    int wm = (wave >> 1) * 32;
    int wn = (wave & 1) * 32;
    int bM = blockIdx.y * 128;
    int bN = blockIdx.x * 64;

    v8f acc[2][2] = {};
    int rowa = lane & 15;
    int hsel = (lane >> 4) << 4;               // K-half select (elements)

    u32 saA = lds_off(&sA[0][0]);
    u32 saB = lds_off(&sB[0][0]);

    auto issue = [&](int k0, int buf) {
#pragma unroll
        for (int p = 0; p < 4; ++p) {          // A tile: 128x64 bf16 = 1024 x 16B
            int g = tid + p * 256;
            const u16* gp = A + (size_t)(bM + (g >> 3)) * lda + k0 + ((g & 7) << 3);
            async_cp16(saA + (u32)buf * (128 * 64 * 2) + (u32)g * 16, gp);
        }
#pragma unroll
        for (int p = 0; p < 2; ++p) {          // B tile: 64x64 bf16 = 512 x 16B
            int g = tid + p * 256;
            const u16* gp = Bw + (size_t)(bN + (g >> 3)) * ldb + k0 + ((g & 7) << 3);
            async_cp16(saB + (u32)buf * (64 * 64 * 2) + (u32)g * 16, gp);
        }
    };

    auto compute = [&](int buf) {
        const u16* pA = &sA[buf][0];
        const u16* pB = &sB[buf][0];
#pragma unroll
        for (int kk = 0; kk < 2; ++kk) {
            int ko = (kk << 5) + hsel;
            v16bf a0 = *(const v16bf*)(pA + (wm      + rowa) * 64 + ko);
            v16bf a1 = *(const v16bf*)(pA + (wm + 16 + rowa) * 64 + ko);
            v16bf b0 = *(const v16bf*)(pB + (wn      + rowa) * 64 + ko);
            v16bf b1 = *(const v16bf*)(pB + (wn + 16 + rowa) * 64 + ko);
            acc[0][0] = __builtin_amdgcn_wmma_f32_16x16x32_bf16(false, a0, false, b0,
                                                                (short)0, acc[0][0], false, false);
            acc[0][1] = __builtin_amdgcn_wmma_f32_16x16x32_bf16(false, a0, false, b1,
                                                                (short)0, acc[0][1], false, false);
            acc[1][0] = __builtin_amdgcn_wmma_f32_16x16x32_bf16(false, a1, false, b0,
                                                                (short)0, acc[1][0], false, false);
            acc[1][1] = __builtin_amdgcn_wmma_f32_16x16x32_bf16(false, a1, false, b1,
                                                                (short)0, acc[1][1], false, false);
        }
    };

    int nK = K >> 6;
    issue(0, 0);
    for (int i = 0; i < nK; ++i) {
        if (i + 1 < nK) {
            issue((i + 1) << 6, (i + 1) & 1);
            asm volatile("s_wait_asynccnt 0x6" ::: "memory");  // stage i resident
        } else {
            asm volatile("s_wait_asynccnt 0x0" ::: "memory");
        }
        __syncthreads();
        compute(i & 1);
        __syncthreads();                        // protect buffer reused by next issue
    }

    // store: vgpr r, lane l -> M = r + 8*(l>>4), N = l&15
    int mrow = ((lane >> 4) << 3);
    int ncol = lane & 15;
#pragma unroll
    for (int i = 0; i < 2; ++i)
#pragma unroll
        for (int j = 0; j < 2; ++j)
#pragma unroll
            for (int r = 0; r < 8; ++r) {
                int row = bM + wm + i * 16 + r + mrow;
                int col = bN + wn + j * 16 + ncol;
                if (row < M && col < N) {
                    size_t o = (size_t)row * ldc + col;
                    float v = acc[i][j][r];
                    C[o] = addC ? (C[o] + v) : v;
                }
            }
}

// ============================================================
// 5) causal depthwise conv (K=4) + SiLU -> f32 (scan u) + bf16 (GEMM A)
// ============================================================
__global__ void conv_silu_kernel(const float* __restrict__ xr,
                                 const float* __restrict__ cw,
                                 const float* __restrict__ cb,
                                 float* __restrict__ xs_f,
                                 u16* __restrict__ xs_bf) {
    int idx = blockIdx.x * 256 + threadIdx.x;
    if (idx >= ROWS * D_INNER) return;
    int d = idx % D_INNER, row = idx / D_INNER;
    int b = row / LQ, l = row % LQ;
    float s = cb[d];
#pragma unroll
    for (int k = 0; k < K_CONV; ++k) {
        int ll = l - (K_CONV - 1) + k;
        if (ll >= 0)
            s = fmaf(cw[d * K_CONV + k],
                     xr[((size_t)(b * LQ + ll)) * (2 * D_INNER) + d], s);
    }
    float v = s * sigm(s);
    xs_f[idx]  = v;
    xs_bf[idx] = f2bf(v);
}

// ============================================================
// 6) delta = softplus(delta_raw + bias)
// ============================================================
__global__ void softplus_bias_kernel(float* __restrict__ delta,
                                     const float* __restrict__ db) {
    int idx = blockIdx.x * 256 + threadIdx.x;
    if (idx >= ROWS * D_INNER) return;
    float v = delta[idx] + db[idx % D_INNER];
    delta[idx] = (v > 20.f) ? v : log1pf(__expf(v));
}

// ============================================================
// 7) selective scan + u*D + silu(res) gating -> bf16 y (GEMM A)
// ============================================================
__global__ __launch_bounds__(256)
void scan_kernel(const float* __restrict__ xs,
                 const float* __restrict__ delta,
                 const float* __restrict__ xdbl,
                 const float* __restrict__ A_log,
                 const float* __restrict__ Dp,
                 const float* __restrict__ xr,
                 u16* __restrict__ y_bf) {
    __shared__ float sBC[32 * 32];
    int tid = threadIdx.x;
    int d = blockIdx.x * 256 + tid;
    int b = blockIdx.y;

    float An[N_STATE];
#pragma unroll
    for (int n = 0; n < N_STATE; ++n) An[n] = -__expf(A_log[d * N_STATE + n]);
    float Dd = Dp[d];
    float st[N_STATE];
#pragma unroll
    for (int n = 0; n < N_STATE; ++n) st[n] = 0.f;

    for (int l0 = 0; l0 < LQ; l0 += 32) {
        __syncthreads();
#pragma unroll
        for (int p = 0; p < 4; ++p) {
            int g = tid + p * 256;
            int step = g >> 5, e = g & 31;
            sBC[g] = xdbl[((size_t)(b * LQ + l0 + step)) * XD + DT_RANK + e];
        }
        __syncthreads();

        for (int s = 0; s < 32; ++s) {
            int l = l0 + s;
            size_t idx = ((size_t)(b * LQ + l)) * D_INNER + d;
            float dt = delta[idx];
            float uu = xs[idx];
            float du = dt * uu;
            const float* Bp = sBC + s * 32;
            const float* Cp = Bp + N_STATE;
            float yv = 0.f;
#pragma unroll
            for (int n = 0; n < N_STATE; ++n) {
                float dA = __expf(dt * An[n]);
                st[n] = fmaf(dA, st[n], du * Bp[n]);
                yv = fmaf(st[n], Cp[n], yv);
            }
            float res = xr[((size_t)(b * LQ + l)) * (2 * D_INNER) + D_INNER + d];
            y_bf[idx] = f2bf((yv + uu * Dd) * (res * sigm(res)));
        }
    }
}

// ============================================================
// 8) final rmsnorm + head (768 -> 7) + de-normalization
// ============================================================
__global__ void final_head_kernel(const float* __restrict__ h,
                                  const float* __restrict__ fnw,
                                  const float* __restrict__ out_w,
                                  const float* __restrict__ mean_s,
                                  const float* __restrict__ std_s,
                                  float* __restrict__ out) {
    __shared__ float red[256];
    __shared__ float scale_s;
    int r = blockIdx.x;
    int b = r / PRED_LEN, t = r % PRED_LEN;
    int l = LQ - PRED_LEN + t;
    int tid = threadIdx.x;
    const float* hr = h + ((size_t)(b * LQ + l)) * D_MODEL;

    float ss = 0.f;
    for (int d = tid; d < D_MODEL; d += 256) { float v = hr[d]; ss += v * v; }
    red[tid] = ss; __syncthreads();
    for (int s = 128; s > 0; s >>= 1) {
        if (tid < s) red[tid] += red[tid + s];
        __syncthreads();
    }
    if (tid == 0) scale_s = rsqrtf(red[0] / (float)D_MODEL + 1e-5f);
    __syncthreads();
    float sc = scale_s;

    for (int c = 0; c < C_OUT; ++c) {
        float p = 0.f;
        for (int d = tid; d < D_MODEL; d += 256)
            p = fmaf(hr[d] * sc * fnw[d], out_w[c * D_MODEL + d], p);
        red[tid] = p; __syncthreads();
        for (int s = 128; s > 0; s >>= 1) {
            if (tid < s) red[tid] += red[tid + s];
            __syncthreads();
        }
        if (tid == 0)
            out[((size_t)(b * PRED_LEN + t)) * C_OUT + c] =
                red[0] * std_s[b * ENC_IN + c] + mean_s[b * ENC_IN + c];
        __syncthreads();
    }
}

// ============================================================
// host launcher
// ============================================================
extern "C" void kernel_launch(void* const* d_in, const int* in_sizes, int n_in,
                              void* d_out, int out_size, void* d_ws, size_t ws_size,
                              hipStream_t stream) {
    const float* x        = (const float*)d_in[0];
    const float* token_w  = (const float*)d_in[1];
    const float* time_w   = (const float*)d_in[2];
    const float* norm_w   = (const float*)d_in[3];
    const float* in_w     = (const float*)d_in[4];
    const float* conv_w   = (const float*)d_in[5];
    const float* conv_b   = (const float*)d_in[6];
    const float* xproj_w  = (const float*)d_in[7];
    const float* dtproj_w = (const float*)d_in[8];
    const float* dtproj_b = (const float*)d_in[9];
    const float* A_log    = (const float*)d_in[10];
    const float* Dp       = (const float*)d_in[11];
    const float* outproj_w= (const float*)d_in[12];
    const float* final_nw = (const float*)d_in[13];
    const float* out_w    = (const float*)d_in[14];
    float* out = (float*)d_out;

    // ---- workspace layout: f32 region, then 16B-aligned bf16 region ----
    float* F = (float*)d_ws;
    float* mean_s = F;                                     // 56
    float* std_s  = F + 56;                                // 56
    float* h      = F + 128;
    float* xr     = h     + (size_t)ROWS * D_MODEL;        // [4096][3072]
    float* xs_f   = xr    + (size_t)ROWS * 2 * D_INNER;    // [4096][1536]
    float* xdbl   = xs_f  + (size_t)ROWS * D_INNER;        // [4096][80]
    float* delta  = xdbl  + (size_t)ROWS * XD;
    u16* U       = (u16*)(delta + (size_t)ROWS * D_INNER);
    u16* wbf_in  = U;                                                  // 2*3072*768
    u16* wbf_xp  = wbf_in  + (size_t)E_LAYERS * 2 * D_INNER * D_MODEL; // 2*128*1536 (padded)
    u16* wbf_dt  = wbf_xp  + (size_t)E_LAYERS * XD_PAD * D_INNER;      // 2*1536*64  (padded)
    u16* wbf_out = wbf_dt  + (size_t)E_LAYERS * D_INNER * DT_PAD;      // 2*768*1536
    u16* xn_bf   = wbf_out + (size_t)E_LAYERS * D_MODEL * D_INNER;
    u16* xs_bf   = xn_bf   + (size_t)ROWS * D_MODEL;
    u16* xdbl_bf = xs_bf   + (size_t)ROWS * D_INNER;
    u16* y_bf    = xdbl_bf + (size_t)ROWS * XD;

    // ---- weight conversions (per launch, deterministic) ----
    {
        int n = E_LAYERS * 2 * D_INNER * D_MODEL;
        cvt_bf16_kernel<<<(n + 255) / 256, 256, 0, stream>>>(in_w, wbf_in, n);
        n = E_LAYERS * D_MODEL * D_INNER;
        cvt_bf16_kernel<<<(n + 255) / 256, 256, 0, stream>>>(outproj_w, wbf_out, n);
        // dt_proj: K 48 -> 64 zero-padded (layer dim folds into rows)
        n = (E_LAYERS * D_INNER) * DT_PAD;
        cvt_pad_bf16_kernel<<<(n + 255) / 256, 256, 0, stream>>>(
            dtproj_w, wbf_dt, E_LAYERS * D_INNER, DT_RANK, E_LAYERS * D_INNER, DT_PAD);
        // x_proj: N 80 -> 128 zero-padded rows (per layer)
        for (int e = 0; e < E_LAYERS; ++e) {
            n = XD_PAD * D_INNER;
            cvt_pad_bf16_kernel<<<(n + 255) / 256, 256, 0, stream>>>(
                xproj_w + (size_t)e * XD * D_INNER,
                wbf_xp  + (size_t)e * XD_PAD * D_INNER,
                XD, D_INNER, XD_PAD, D_INNER);
        }
    }

    stats_kernel<<<BQ * ENC_IN, 256, 0, stream>>>(x, mean_s, std_s);
    embed_kernel<<<(ROWS * D_MODEL + 255) / 256, 256, 0, stream>>>(
        x, token_w, time_w, mean_s, std_s, h);

    for (int e = 0; e < E_LAYERS; ++e) {
        const float* nw  = norm_w   + (size_t)e * D_MODEL;
        const float* cw  = conv_w   + (size_t)e * D_INNER * K_CONV;
        const float* cb  = conv_b   + (size_t)e * D_INNER;
        const float* dtb = dtproj_b + (size_t)e * D_INNER;
        const float* al  = A_log    + (size_t)e * D_INNER * N_STATE;
        const float* dp  = Dp       + (size_t)e * D_INNER;
        const u16* iw  = wbf_in  + (size_t)e * 2 * D_INNER * D_MODEL;
        const u16* xpw = wbf_xp  + (size_t)e * XD_PAD * D_INNER;
        const u16* dtw = wbf_dt  + (size_t)e * D_INNER * DT_PAD;
        const u16* ow  = wbf_out + (size_t)e * D_MODEL * D_INNER;

        rmsnorm_kernel<<<ROWS, 256, 0, stream>>>(h, nw, xn_bf);

        {   // in_proj: [4096,768] x [3072,768]^T -> xr
            dim3 g((2 * D_INNER + 63) / 64, ROWS / 128);
            gemm_bf16_wmma<<<g, 256, 0, stream>>>(xn_bf, D_MODEL, iw, D_MODEL,
                                                  xr, 2 * D_INNER,
                                                  ROWS, 2 * D_INNER, D_MODEL, 0);
        }
        conv_silu_kernel<<<(ROWS * D_INNER + 255) / 256, 256, 0, stream>>>(
            xr, cw, cb, xs_f, xs_bf);

        {   // x_proj: [4096,1536] x [80(pad 128),1536]^T -> xdbl [4096,80]
            dim3 g((XD + 63) / 64, ROWS / 128);
            gemm_bf16_wmma<<<g, 256, 0, stream>>>(xs_bf, D_INNER, xpw, D_INNER,
                                                  xdbl, XD,
                                                  ROWS, XD, D_INNER, 0);
        }
        {   // x_dbl -> bf16 (delta GEMM input)
            int n = ROWS * XD;
            cvt_bf16_kernel<<<(n + 255) / 256, 256, 0, stream>>>(xdbl, xdbl_bf, n);
        }
        {   // dt_proj: xdbl[:, :64] x [1536, 64(zero-padded 48:64)]^T -> delta
            dim3 g((D_INNER + 63) / 64, ROWS / 128);
            gemm_bf16_wmma<<<g, 256, 0, stream>>>(xdbl_bf, XD, dtw, DT_PAD,
                                                  delta, D_INNER,
                                                  ROWS, D_INNER, DT_PAD, 0);
        }
        softplus_bias_kernel<<<(ROWS * D_INNER + 255) / 256, 256, 0, stream>>>(delta, dtb);

        {   // selective scan + gate -> y_bf
            dim3 g(D_INNER / 256, BQ);
            scan_kernel<<<g, 256, 0, stream>>>(xs_f, delta, xdbl, al, dp, xr, y_bf);
        }
        {   // out_proj with residual accumulate into h
            dim3 g((D_MODEL + 63) / 64, ROWS / 128);
            gemm_bf16_wmma<<<g, 256, 0, stream>>>(y_bf, D_INNER, ow, D_INNER,
                                                  h, D_MODEL,
                                                  ROWS, D_MODEL, D_INNER, 1);
        }
    }

    final_head_kernel<<<BQ * PRED_LEN, 256, 0, stream>>>(h, final_nw, out_w,
                                                         mean_s, std_s, out);
}
